// LSTM_57758720197117
// MI455X (gfx1250) — matrix operand order repaired and checked by reading, not compiled, not run on hardware
//
#include <hip/hip_runtime.h>
#include <math.h>

// ---------------- types ----------------
typedef __attribute__((ext_vector_type(16))) __bf16 bf16x16;
typedef __attribute__((ext_vector_type(8)))  float  f32x8;
typedef __attribute__((ext_vector_type(4)))  unsigned int u32x4;

union Frag16 {           // 16 bf16 = 32 bytes = 8 VGPRs (wave32 A/B operand)
    bf16x16 v;
    u32x4   q[2];
};

// ---------------- constants ----------------
#define T_STEPS 1024
#define BATCH   128
#define HID     200
#define NPAD    208      // per-gate padded width (13 tiles of 16)
#define GROWS   832      // 4*NPAD
#define HSTR    224      // h / y row stride (7 k-steps of 32)
#define NWG     13       // persistent workgroups: 13*8 waves = 104 = 8 mtiles * 13 jtiles

__device__ __forceinline__ unsigned short f2bf(float x) {
    unsigned u = __float_as_uint(x);
    unsigned r = u + 0x7FFFu + ((u >> 16) & 1u);   // round-to-nearest-even
    return (unsigned short)(r >> 16);
}
__device__ __forceinline__ float bf2f_lo(unsigned u) { return __uint_as_float(u << 16); }
__device__ __forceinline__ float bf2f_hi(unsigned u) { return __uint_as_float(u & 0xffff0000u); }
__device__ __forceinline__ float sigm(float x) { return 1.0f / (1.0f + __expf(-x)); }

// ---------------- prep kernels ----------------
__global__ void k_zero(unsigned* __restrict__ p, size_t n) {
    size_t i = (size_t)blockIdx.x * blockDim.x + threadIdx.x;
    size_t s = (size_t)gridDim.x * blockDim.x;
    for (; i < n; i += s) p[i] = 0u;
}

// inp [T*B][100] f32 -> [T*B][128] bf16 zero-padded
__global__ void k_cvt_inp(const float* __restrict__ inp, unsigned short* __restrict__ dst) {
    size_t n = (size_t)T_STEPS * BATCH * 128;
    size_t i = (size_t)blockIdx.x * blockDim.x + threadIdx.x;
    size_t s = (size_t)gridDim.x * blockDim.x;
    for (; i < n; i += s) {
        size_t row = i >> 7;
        int k = (int)(i & 127);
        dst[i] = (k < 100) ? f2bf(inp[row * 100 + k]) : (unsigned short)0;
    }
}

// Padded bf16 weight matrix [832][kpad] from src [800][kreal] (gate-major rows).
__global__ void k_build_w(const float* __restrict__ W, unsigned short* __restrict__ Wc,
                          int kreal, int kpad) {
    size_t n = (size_t)GROWS * kpad;
    size_t i = (size_t)blockIdx.x * blockDim.x + threadIdx.x;
    size_t s = (size_t)gridDim.x * blockDim.x;
    for (; i < n; i += s) {
        int r = (int)(i / kpad), k = (int)(i % kpad);
        int g = r / NPAD, j = r % NPAD;
        float v = (j < HID && k < kreal) ? W[(size_t)(g * HID + j) * kreal + k] : 0.0f;
        Wc[i] = f2bf(v);
    }
}

__global__ void k_build_bias(const float* __restrict__ bih, const float* __restrict__ bhh,
                             float* __restrict__ bc) {
    int i = blockIdx.x * blockDim.x + threadIdx.x;
    if (i >= GROWS) return;
    int g = i / NPAD, j = i % NPAD;
    bc[i] = (j < HID) ? (bih[g * HID + j] + bhh[g * HID + j]) : 0.0f;
}

// fc_W [100][200] -> [112][224] bf16 padded; fc_b -> [112] f32 padded
__global__ void k_build_fc(const float* __restrict__ fcW, const float* __restrict__ fcb,
                           unsigned short* __restrict__ Wp, float* __restrict__ bp) {
    int i = blockIdx.x * blockDim.x + threadIdx.x;
    if (i >= 112 * 224) return;
    int n = i / 224, k = i % 224;
    float v = (n < 100 && k < HID) ? fcW[n * HID + k] : 0.0f;
    Wp[i] = f2bf(v);
    if (k == 0) bp[n] = (n < 100) ? fcb[n] : 0.0f;
}

// h0 [2][128][200] f32 -> hgA bf16 [128][224] padded; hgB zeroed
__global__ void k_init_h(const float* __restrict__ h0,
                         unsigned short* __restrict__ hgA, unsigned short* __restrict__ hgB,
                         int layer) {
    int i = blockIdx.x * blockDim.x + threadIdx.x;
    if (i >= BATCH * HSTR) return;
    int b = i / HSTR, j = i % HSTR;
    hgA[i] = (j < HID) ? f2bf(h0[(size_t)layer * BATCH * HID + b * HID + j]) : (unsigned short)0;
    hgB[i] = 0;
}

// ---------------- bulk x-projection GEMM ----------------
// Gx[t][n][b] (bf16, transposed) = X[t*B+b][:] . Wx[n][:]
// grid = dim3(52, 1024), block = 256 (8 waves, one 16x16 tile each)
__global__ __launch_bounds__(256) void k_xproj(
    const unsigned short* __restrict__ X,    // [T*B][xstr] bf16 zero-padded
    const unsigned short* __restrict__ Wx,   // [832][kpad] bf16
    unsigned short* __restrict__ Gx,         // [T][832][128] bf16
    int xstr, int kst)                        // kpad == xstr; kst = xstr/32
{
    const int lane  = threadIdx.x & 31;
    const int wave  = threadIdx.x >> 5;
    const int ntile = blockIdx.x;                 // 0..51
    const int mtile = blockIdx.y * 8 + wave;      // 0..8191
    const int nl = lane & 15, hi = lane >> 4;
    const int n  = ntile * 16 + nl;               // 0..831
    const int m  = mtile * 16 + nl;

    f32x8 acc;
    #pragma unroll
    for (int r = 0; r < 8; ++r) acc[r] = 0.0f;

    const unsigned short* arow = X + (size_t)m * xstr;
    const unsigned short* wrow = Wx + (size_t)n * xstr;

    for (int kk = 0; kk < kst; ++kk) {
        Frag16 af, bfv;
        const int kb = kk * 32 + hi * 16;
        af.q[0]  = *(const u32x4*)(arow + kb);
        af.q[1]  = *(const u32x4*)(arow + kb + 8);
        const unsigned short* wr = wrow + kk * 32 + hi * 16;
        bfv.q[0] = *(const u32x4*)(wr);
        bfv.q[1] = *(const u32x4*)(wr + 8);
        acc = __builtin_amdgcn_wmma_f32_16x16x32_bf16(
            false, af.v, false, bfv.v, (short)0, acc, false, false);
    }

    // transposed packed store: 8 consecutive batch entries -> one 16B store
    const int t    = mtile >> 3;                  // (mtile*16)/128
    const int brow = (mtile & 7) * 16 + hi * 8;
    u32x4 pk;
    #pragma unroll
    for (int p = 0; p < 4; ++p)
        pk[p] = (unsigned)f2bf(acc[2 * p]) | ((unsigned)f2bf(acc[2 * p + 1]) << 16);
    *(u32x4*)(Gx + ((size_t)t * GROWS + n) * BATCH + brow) = pk;
}

// ---------------- grid barrier ----------------
__device__ __forceinline__ void grid_barrier(unsigned* bar, unsigned expected) {
    __threadfence();
    __syncthreads();
    if (threadIdx.x == 0) {
        __hip_atomic_fetch_add(bar, 1u, __ATOMIC_RELEASE, __HIP_MEMORY_SCOPE_AGENT);
        while (__hip_atomic_load(bar, __ATOMIC_ACQUIRE, __HIP_MEMORY_SCOPE_AGENT) < expected)
            __builtin_amdgcn_s_sleep(2);
    }
    __syncthreads();
    __threadfence();
}

// ---------------- persistent single-layer LSTM recurrence ----------------
// Per step: acc = bias + Gx[t] (4 b128 loads), 7 k-steps of h-projection with
// W_hh fragments held in registers (4*7*8 = 224 VGPRs), fused gates, barrier.
__global__ __launch_bounds__(256) void k_lstm_layer(
    const unsigned short* __restrict__ Gx,     // [T][832][128] bf16
    const unsigned short* __restrict__ Wh,     // [832][224] bf16
    const float* __restrict__ bc,              // [832]
    unsigned short* __restrict__ hgA, unsigned short* __restrict__ hgB,
    unsigned short* __restrict__ yout,         // [T*B][224] bf16
    const float* __restrict__ c0in,            // [128][200] layer slice
    float* __restrict__ hn_out,                // [128][200] layer slice
    float* __restrict__ cn_out,                // [128][200] layer slice
    unsigned* __restrict__ bar)
{
    const int tid   = threadIdx.x;
    const int lane  = tid & 31;
    const int wid   = blockIdx.x * 8 + (tid >> 5);     // 0..103
    const int mtile = wid & 7;                          // 0..7
    const int jtile = wid >> 3;                         // 0..12
    const int nl    = lane & 15;
    const int hi    = lane >> 4;                        // 0/1
    const int jcol  = jtile * 16 + nl;                  // 0..207 gate-local column
    const int m     = mtile * 16 + nl;                  // A-row (batch) for this lane
    const int brow  = mtile * 16 + hi * 8;              // first accumulator batch row

    // ---- preload all W_hh fragments into registers (persistent across t) ----
    Frag16 whf[4][7];
    #pragma unroll
    for (int g = 0; g < 4; ++g) {
        const unsigned short* wr0 = Wh + (size_t)(g * NPAD + jcol) * HSTR;
        #pragma unroll
        for (int kk = 0; kk < 7; ++kk) {
            whf[g][kk].q[0] = *(const u32x4*)(wr0 + kk * 32 + hi * 16);
            whf[g][kk].q[1] = *(const u32x4*)(wr0 + kk * 32 + hi * 16 + 8);
        }
    }

    // ---- persistent cell state: 8 batches x fixed column jcol ----
    float cst[8];
    #pragma unroll
    for (int r = 0; r < 8; ++r)
        cst[r] = (jcol < HID) ? c0in[(brow + r) * HID + jcol] : 0.0f;

    float bias[4];
    #pragma unroll
    for (int g = 0; g < 4; ++g) bias[g] = bc[g * NPAD + jcol];

    for (int t = 0; t < T_STEPS; ++t) {
        const unsigned short* hgR = (t & 1) ? hgB : hgA;   // read h_{t-1}
        unsigned short*       hgW = (t & 1) ? hgA : hgB;   // write h_t

        // acc init: bias + precomputed x-projection (transposed layout)
        f32x8 acc[4];
        const unsigned short* gxt = Gx + (size_t)t * GROWS * BATCH;
        #pragma unroll
        for (int g = 0; g < 4; ++g) {
            u32x4 q = *(const u32x4*)(gxt + (size_t)(g * NPAD + jcol) * BATCH + brow);
            #pragma unroll
            for (int p = 0; p < 4; ++p) {
                acc[g][2 * p]     = bias[g] + bf2f_lo(q[p]);
                acc[g][2 * p + 1] = bias[g] + bf2f_hi(q[p]);
            }
        }
        if (t + 1 < T_STEPS) {
            const unsigned short* gxn = Gx + (size_t)(t + 1) * GROWS * BATCH;
            #pragma unroll
            for (int g = 0; g < 4; ++g)
                __builtin_prefetch(gxn + (size_t)(g * NPAD + jcol) * BATCH + brow, 0, 0);
        }

        // ---- h-projection: 7 k-steps, weights already in registers ----
        const unsigned short* hrow = hgR + m * HSTR;
        #pragma unroll
        for (int kk = 0; kk < 7; ++kk) {
            Frag16 af;
            const int kb = kk * 32 + hi * 16;
            af.q[0] = *(const u32x4*)(hrow + kb);
            af.q[1] = *(const u32x4*)(hrow + kb + 8);
            #pragma unroll
            for (int g = 0; g < 4; ++g)
                acc[g] = __builtin_amdgcn_wmma_f32_16x16x32_bf16(
                    false, af.v, false, whf[g][kk].v, (short)0, acc[g], false, false);
        }

        // ---- fused LSTM elementwise (all four gates in registers) ----
        #pragma unroll
        for (int r = 0; r < 8; ++r) {
            float ig = sigm(acc[0][r]);
            float fg = sigm(acc[1][r]);
            float gg = tanhf(acc[2][r]);
            float og = sigm(acc[3][r]);
            float c  = fg * cst[r] + ig * gg;
            cst[r]   = c;
            float h  = og * tanhf(c);
            if (jcol < HID) {
                int b = brow + r;
                unsigned short hb = f2bf(h);
                hgW[b * HSTR + jcol] = hb;
                yout[((size_t)t * BATCH + b) * HSTR + jcol] = hb;
                if (t == T_STEPS - 1) {
                    hn_out[b * HID + jcol] = h;
                    cn_out[b * HID + jcol] = c;
                }
            }
        }

        grid_barrier(bar, (unsigned)(t + 1) * NWG);
    }
}

// ---------------- head GEMM: out = y1 @ fc_W^T + fc_b ----------------
__global__ __launch_bounds__(256) void k_head(
    const unsigned short* __restrict__ y1,   // [T*B][224] bf16 (zero padded)
    const unsigned short* __restrict__ Wfc,  // [112][224] bf16
    const float* __restrict__ bfc,           // [112]
    float* __restrict__ out)                 // [T*B][100]
{
    const int lane  = threadIdx.x & 31;
    const int wave  = threadIdx.x >> 5;
    const int ntile = blockIdx.x;                 // 0..6
    const int mtile = blockIdx.y * 8 + wave;      // 0..8191
    const int nl = lane & 15, hi = lane >> 4;
    const int n  = ntile * 16 + nl;               // 0..111
    const int m  = mtile * 16 + nl;

    f32x8 acc;
    {
        float bv = bfc[n];
        #pragma unroll
        for (int r = 0; r < 8; ++r) acc[r] = bv;
    }
    const unsigned short* arow = y1 + (size_t)m * HSTR;
    const unsigned short* wrow = Wfc + (size_t)n * HSTR;

    #pragma unroll
    for (int kk = 0; kk < 7; ++kk) {
        Frag16 af, bfv;
        const int kb = kk * 32 + hi * 16;
        af.q[0]  = *(const u32x4*)(arow + kb);
        af.q[1]  = *(const u32x4*)(arow + kb + 8);
        const unsigned short* wr = wrow + kk * 32 + hi * 16;
        bfv.q[0] = *(const u32x4*)(wr);
        bfv.q[1] = *(const u32x4*)(wr + 8);
        acc = __builtin_amdgcn_wmma_f32_16x16x32_bf16(
            false, af.v, false, bfv.v, (short)0, acc, false, false);
    }
    if (n < 100) {
        #pragma unroll
        for (int r = 0; r < 8; ++r) {
            int mg = mtile * 16 + r + hi * 8;
            out[(size_t)mg * 100 + n] = acc[r];
        }
    }
}

// ---------------- launch ----------------
extern "C" void kernel_launch(void* const* d_in, const int* in_sizes, int n_in,
                              void* d_out, int out_size, void* d_ws, size_t ws_size,
                              hipStream_t stream) {
    (void)in_sizes; (void)n_in; (void)out_size; (void)ws_size;
    const float* inp  = (const float*)d_in[0];
    const float* h0   = (const float*)d_in[1];
    const float* c0   = (const float*)d_in[2];
    const float* Wih0 = (const float*)d_in[3];
    const float* Whh0 = (const float*)d_in[4];
    const float* bih0 = (const float*)d_in[5];
    const float* bhh0 = (const float*)d_in[6];
    const float* Wih1 = (const float*)d_in[7];
    const float* Whh1 = (const float*)d_in[8];
    const float* bih1 = (const float*)d_in[9];
    const float* bhh1 = (const float*)d_in[10];
    const float* fcW  = (const float*)d_in[11];
    const float* fcb  = (const float*)d_in[12];

    float* out = (float*)d_out;
    float* hn  = out + (size_t)T_STEPS * BATCH * 100;
    float* cn  = hn + 2 * BATCH * HID;

    char* ws = (char*)d_ws;
    size_t off = 0;
    auto take = [&](size_t bytes) -> char* {
        char* p = ws + off;
        off = (off + bytes + 255) & ~(size_t)255;
        return p;
    };
    unsigned*       bar   = (unsigned*)      take(256);
    unsigned short* inp16 = (unsigned short*)take((size_t)T_STEPS * BATCH * 128 * 2);
    unsigned short* wx0   = (unsigned short*)take((size_t)GROWS * 128 * 2);
    unsigned short* wh0   = (unsigned short*)take((size_t)GROWS * HSTR * 2);
    unsigned short* wx1   = (unsigned short*)take((size_t)GROWS * HSTR * 2);
    unsigned short* wh1   = (unsigned short*)take((size_t)GROWS * HSTR * 2);
    float*          bc0   = (float*)         take(GROWS * 4);
    float*          bc1   = (float*)         take(GROWS * 4);
    unsigned short* fcw   = (unsigned short*)take(112 * 224 * 2);
    float*          fcbv  = (float*)         take(112 * 4);
    unsigned short* hg0A  = (unsigned short*)take(BATCH * HSTR * 2);
    unsigned short* hg0B  = (unsigned short*)take(BATCH * HSTR * 2);
    unsigned short* hg1A  = (unsigned short*)take(BATCH * HSTR * 2);
    unsigned short* hg1B  = (unsigned short*)take(BATCH * HSTR * 2);
    unsigned short* y0b   = (unsigned short*)take((size_t)T_STEPS * BATCH * HSTR * 2);
    unsigned short* y1b   = (unsigned short*)take((size_t)T_STEPS * BATCH * HSTR * 2);
    unsigned short* gx    = (unsigned short*)take((size_t)T_STEPS * GROWS * BATCH * 2); // 218 MB, reused by both layers

    // ---- init / conversion ----
    k_zero<<<64, 256, 0, stream>>>(bar, 64);
    k_zero<<<2048, 256, 0, stream>>>((unsigned*)y0b, (size_t)T_STEPS * BATCH * HSTR / 2);
    k_zero<<<2048, 256, 0, stream>>>((unsigned*)y1b, (size_t)T_STEPS * BATCH * HSTR / 2);
    k_cvt_inp<<<4096, 256, 0, stream>>>(inp, inp16);
    k_build_w<<<416, 256, 0, stream>>>(Wih0, wx0, 100, 128);
    k_build_w<<<728, 256, 0, stream>>>(Whh0, wh0, 200, HSTR);
    k_build_w<<<728, 256, 0, stream>>>(Wih1, wx1, 200, HSTR);
    k_build_w<<<728, 256, 0, stream>>>(Whh1, wh1, 200, HSTR);
    k_build_bias<<<4, 256, 0, stream>>>(bih0, bhh0, bc0);
    k_build_bias<<<4, 256, 0, stream>>>(bih1, bhh1, bc1);
    k_build_fc<<<98, 256, 0, stream>>>(fcW, fcb, fcw, fcbv);
    k_init_h<<<112, 256, 0, stream>>>(h0, hg0A, hg0B, 0);
    k_init_h<<<112, 256, 0, stream>>>(h0, hg1A, hg1B, 1);

    // ---- layer 0: bulk x-projection, then persistent recurrence ----
    k_xproj<<<dim3(52, 1024, 1), dim3(256, 1, 1), 0, stream>>>(inp16, wx0, gx, 128, 4);
    k_lstm_layer<<<NWG, 256, 0, stream>>>(gx, wh0, bc0, hg0A, hg0B, y0b,
                                          c0, hn, cn, bar);

    // ---- layer 1: bulk x-projection on y0, then persistent recurrence ----
    k_zero<<<64, 256, 0, stream>>>(bar, 64);
    k_xproj<<<dim3(52, 1024, 1), dim3(256, 1, 1), 0, stream>>>(y0b, wx1, gx, HSTR, 7);
    k_lstm_layer<<<NWG, 256, 0, stream>>>(gx, wh1, bc1, hg1A, hg1B, y1b,
                                          c0 + (size_t)BATCH * HID,
                                          hn + (size_t)BATCH * HID,
                                          cn + (size_t)BATCH * HID, bar);

    // ---- linear head ----
    k_head<<<dim3(7, 1024, 1), dim3(256, 1, 1), 0, stream>>>(y1b, fcw, fcbv, out);
}
